// DiffAttn_76759655514265
// MI455X (gfx1250) — matrix operand order repaired and checked by reading, not compiled, not run on hardware
//
#include <hip/hip_runtime.h>
#include <hip/hip_bf16.h>
#include <math.h>

// ---------------------------------------------------------------------------
// DiffAttn for MI455X (gfx1250, wave32, WMMA bf16 16x16x32)
//   B=2, S=4096, D_IN=1024, d=64 (2d = 128)
// ---------------------------------------------------------------------------

typedef __attribute__((ext_vector_type(16))) __bf16       v16bf;
typedef __attribute__((ext_vector_type(8)))  float        v8f;
typedef __attribute__((ext_vector_type(4)))  float        f32x4;
typedef __attribute__((ext_vector_type(4)))  unsigned int u32x4;

union ABf16 {
    v16bf  v;
    __bf16 e[16];
    unsigned int u[8];
    u32x4  u4[2];
};

__device__ __forceinline__ v8f wmma_bf16(const ABf16& a, const ABf16& b, v8f c) {
    return __builtin_amdgcn_wmma_f32_16x16x32_bf16(false, a.v, false, b.v,
                                                   (short)0, c, false, false);
}

// ---------------------------------------------------------------------------
// Kernel 0: pack W{q,k,v}[1024][128] f32 into column-major bf16
//   Wt[384][1024]  (n = 0..127 -> Wq cols, 128..255 -> Wk, 256..383 -> Wv)
// One-time 1.5MB -> 768KB pre-pass so the projection's B-operand loads
// become aligned b128 pair loads.
// ---------------------------------------------------------------------------
__global__ __launch_bounds__(256) void DiffAttn_packW(
        const float* __restrict__ Wq,
        const float* __restrict__ Wk,
        const float* __restrict__ Wv,
        __bf16* __restrict__ Wt)
{
    const int idx = (int)(blockIdx.x * 256 + threadIdx.x);   // 384*1024 total
    const int n = idx >> 10;
    const int d = idx & 1023;
    const float* __restrict__ W = (n < 128) ? Wq : (n < 256) ? Wk : Wv;
    Wt[(size_t)n * 1024 + d] = (__bf16)W[(size_t)d * 128 + (n & 127)];
}

// ---------------------------------------------------------------------------
// Kernel 1: fused QKV projection.
//   X[8192][1024] f32  @  Wt[384][1024] bf16 (col-major weights)
//   -> Qb[8192][128] bf16 row major (pre-scaled by 1/sqrt(d)*log2(e)),
//      Kb[8192][128] bf16 row major,
//      Vt[2][128][4096] bf16 (V transposed per batch)
// One wave computes one 16x16 output tile; 32 x wmma_f32_16x16x32_bf16.
// ---------------------------------------------------------------------------
__global__ __launch_bounds__(256) void DiffAttn_qkv_proj(
        const float* __restrict__ X,
        const __bf16* __restrict__ Wt,
        __bf16* __restrict__ Qb,
        __bf16* __restrict__ Kb,
        __bf16* __restrict__ Vt)
{
    const int wave = (int)((blockIdx.x * blockDim.x + threadIdx.x) >> 5);
    const int lane = (int)(threadIdx.x & 31);
    const int NT = 24;                       // 384 / 16 column tiles
    if (wave >= 512 * NT) return;
    const int mt = wave / NT;
    const int nt = wave % NT;
    const int m0 = mt * 16;
    const int nglob = nt * 16;
    const int sect = nglob >> 7;             // 0=Q 1=K 2=V
    const int ncol = nglob & 127;

    const int q = lane & 15;                 // A: row within tile / B: col
    const int h = lane >> 4;

    // A-operand: d = dbase + 8h + {0..7, 16..23}  (8h folded into base, 32B align)
    const float* __restrict__ xbase = X + (size_t)(m0 + q) * 1024 + 8 * h;
    // B-operand: d = dbase + 8h + {0..7, 16..23} -> u32x4 idx = dbase/8 + h + {0,2}
    const u32x4* __restrict__ wbase4 =
        (const u32x4*)(Wt + (size_t)(nglob + q) * 1024) + h;

    v8f c = {};
    for (int dbase = 0; dbase < 1024; dbase += 32) {
        ABf16 a, b;
        const f32x4* __restrict__ xr4 = (const f32x4*)(xbase + dbase);
        const f32x4 x0 = xr4[0];             // d 0..3
        const f32x4 x1 = xr4[1];             // d 4..7
        const f32x4 x2 = xr4[4];             // d 16..19
        const f32x4 x3 = xr4[5];             // d 20..23
#pragma unroll
        for (int i = 0; i < 4; ++i) {
            a.e[i]      = (__bf16)x0[i];
            a.e[4 + i]  = (__bf16)x1[i];
            a.e[8 + i]  = (__bf16)x2[i];
            a.e[12 + i] = (__bf16)x3[i];
        }
        b.u4[0] = wbase4[(dbase >> 3)];
        b.u4[1] = wbase4[(dbase >> 3) + 2];
        c = wmma_bf16(a, b, c);
    }

    if (sect == 0) {
        // Q: pre-scale by 1/sqrt(64)*log2(e) so flash kernel needs no S scaling
        const float qsc = 0.125f * 1.44269504088896340736f;
#pragma unroll
        for (int r = 0; r < 8; ++r) {        // C layout: row m = r + 8h, col = q
            const int m = m0 + r + 8 * h;
            Qb[(size_t)m * 128 + ncol + q] = (__bf16)(c[r] * qsc);
        }
    } else if (sect == 1) {
#pragma unroll
        for (int r = 0; r < 8; ++r) {
            const int m = m0 + r + 8 * h;
            Kb[(size_t)m * 128 + ncol + q] = (__bf16)c[r];
        }
    } else {
        // V transposed: Vt[b][col][s]; 8 adjacent rows -> one b128 store
        const int bb = m0 >> 12;
        const int s0 = m0 & 4095;
        __bf16* __restrict__ vp =
            Vt + (size_t)(bb * 128 + ncol + q) * 4096 + s0 + 8 * h;
#pragma unroll
        for (int i = 0; i < 8; ++i) vp[i] = (__bf16)c[i];
    }
}

// ---------------------------------------------------------------------------
// Kernel 2: flash differential attention.
// One wave handles 16 queries x 64 V-columns. Two online-softmax branches.
// S^T = K_tile @ Q^T trick: exp'd logits in C-layout ARE the A-operand layout
// of P for the P@V WMMA (no cross-lane movement). Softmax in log2 domain
// (scale pre-folded into Q). All operand fetches are aligned b128 loads.
// ---------------------------------------------------------------------------
__global__ __launch_bounds__(256) void DiffAttn_flash(
        const __bf16* __restrict__ Qb,
        const __bf16* __restrict__ Kb,
        const __bf16* __restrict__ Vt,
        const float* __restrict__ lam_p,
        float* __restrict__ Out)
{
    const int wave = (int)((blockIdx.x * blockDim.x + threadIdx.x) >> 5);
    const int lane = (int)(threadIdx.x & 31);
    if (wave >= 1024) return;
    const int q  = lane & 15;
    const int h  = lane >> 4;
    const int bb = wave >> 9;                // batch
    const int qt = (wave >> 1) & 255;        // query tile
    const int vh = wave & 1;                 // which 64-col half of V
    const int q0 = qt * 16;
    const size_t rowbase = (size_t)bb * 4096;
    const float lam = *lam_p;

    // Q as B-operand of S^T = K @ Q^T : lane col = query, K-dim = d.
    // u32 index base 16cc + 8h  ->  u32x4 index 4cc + 2h (+1); branch2 +8.
    ABf16 qb1[2], qb2[2];
    {
        const u32x4* __restrict__ Qp4 =
            (const u32x4*)(Qb + (rowbase + q0 + q) * 128);
#pragma unroll
        for (int cc = 0; cc < 2; ++cc) {
            qb1[cc].u4[0] = Qp4[4 * cc + 2 * h];
            qb1[cc].u4[1] = Qp4[4 * cc + 2 * h + 1];
            qb2[cc].u4[0] = Qp4[4 * cc + 2 * h + 8];
            qb2[cc].u4[1] = Qp4[4 * cc + 2 * h + 9];
        }
    }

    float m1 = -INFINITY, m2 = -INFINITY, l1 = 0.f, l2 = 0.f;
    v8f o1[4], o2[4];
#pragma unroll
    for (int v = 0; v < 4; ++v)
#pragma unroll
        for (int r = 0; r < 8; ++r) { o1[v][r] = 0.f; o2[v][r] = 0.f; }

    for (int k0 = 0; k0 < 4096; k0 += 32) {
        if (k0 + 32 < 4096) {
            __builtin_prefetch(Kb + (rowbase + k0 + 32 + q) * 128, 0, 1);
            __builtin_prefetch(Vt + ((size_t)(bb * 128 + vh * 64 + q) * 4096
                                     + (size_t)(k0 + 32)), 0, 1);
        }

        // ---- logits S^T (keys x queries) for 32 keys, both branches ----
        v8f s1[2], s2[2];
#pragma unroll
        for (int t = 0; t < 2; ++t) {
            // A-operand (keys x d): u32x4 idx 4cc + 2h and 4cc + 2h + 1; br2 +8.
            const u32x4* __restrict__ Kp4 =
                (const u32x4*)(Kb + (rowbase + k0 + 16 * t + q) * 128);
            ABf16 ka1[2], ka2[2];
#pragma unroll
            for (int cc = 0; cc < 2; ++cc) {
                ka1[cc].u4[0] = Kp4[4 * cc + 2 * h];
                ka1[cc].u4[1] = Kp4[4 * cc + 2 * h + 1];
                ka2[cc].u4[0] = Kp4[4 * cc + 2 * h + 8];
                ka2[cc].u4[1] = Kp4[4 * cc + 2 * h + 9];
            }
            v8f z = {};
            v8f a1 = wmma_bf16(ka1[0], qb1[0], z);
            s1[t]  = wmma_bf16(ka1[1], qb1[1], a1);
            v8f a2 = wmma_bf16(ka2[0], qb2[0], z);
            s2[t]  = wmma_bf16(ka2[1], qb2[1], a2);
        }

        // ---- online softmax per query (lane owns query q, 16 keys in-lane).
        //      Logits already in log2 domain (scale folded into Q). ----
        float mx1 = -INFINITY, mx2 = -INFINITY;
#pragma unroll
        for (int t = 0; t < 2; ++t)
#pragma unroll
            for (int r = 0; r < 8; ++r) {
                mx1 = fmaxf(mx1, s1[t][r]);
                mx2 = fmaxf(mx2, s2[t][r]);
            }
        mx1 = fmaxf(mx1, __shfl_xor(mx1, 16, 32));
        mx2 = fmaxf(mx2, __shfl_xor(mx2, 16, 32));
        const float m1n = fmaxf(m1, mx1);
        const float m2n = fmaxf(m2, mx2);
        const float f1 = __builtin_amdgcn_exp2f(m1 - m1n);
        const float f2 = __builtin_amdgcn_exp2f(m2 - m2n);

        float rs1 = 0.f, rs2 = 0.f;
#pragma unroll
        for (int t = 0; t < 2; ++t)
#pragma unroll
            for (int r = 0; r < 8; ++r) {
                const float e1 = __builtin_amdgcn_exp2f(s1[t][r] - m1n);
                const float e2 = __builtin_amdgcn_exp2f(s2[t][r] - m2n);
                s1[t][r] = e1; s2[t][r] = e2;
                rs1 += e1; rs2 += e2;
            }
        rs1 += __shfl_xor(rs1, 16, 32);
        rs2 += __shfl_xor(rs2, 16, 32);
        l1 = l1 * f1 + rs1;  m1 = m1n;
        l2 = l2 * f2 + rs2;  m2 = m2n;

        // ---- rescale accumulators (row m = r+8h -> stats live in lane r+8h) ----
#pragma unroll
        for (int r = 0; r < 8; ++r) {
            const float g1 = __shfl(f1, r + 8 * h, 32);
            const float g2 = __shfl(f2, r + 8 * h, 32);
#pragma unroll
            for (int v = 0; v < 4; ++v) { o1[v][r] *= g1; o2[v][r] *= g2; }
        }

        // ---- P (16q x 32k) A-operand == concat of the two S^T C-tiles ----
        ABf16 p1, p2;
#pragma unroll
        for (int i = 0; i < 8; ++i) {
            p1.e[i]     = (__bf16)s1[0][i];
            p1.e[i + 8] = (__bf16)s1[1][i];
            p2.e[i]     = (__bf16)s2[0][i];
            p2.e[i + 8] = (__bf16)s2[1][i];
        }

        // ---- O += P @ V over 4 column tiles (two b128 loads per tile) ----
#pragma unroll
        for (int v = 0; v < 4; ++v) {
            const int col = vh * 64 + v * 16 + q;
            const u32x4* __restrict__ Vp4 =
                (const u32x4*)(Vt + (size_t)(bb * 128 + col) * 4096 + k0);
            ABf16 bv;
            bv.u4[0] = Vp4[h];
            bv.u4[1] = Vp4[h + 2];
            o1[v] = wmma_bf16(p1, bv, o1[v]);
            o2[v] = wmma_bf16(p2, bv, o2[v]);
        }
    }

    // ---- combine branches: O1/l1 - lam * O2/l2, store f32 ----
#pragma unroll
    for (int r = 0; r < 8; ++r) {
        const float il1 = 1.0f / __shfl(l1, r + 8 * h, 32);
        const float il2 = 1.0f / __shfl(l2, r + 8 * h, 32);
        const int mrow = q0 + r + 8 * h;
#pragma unroll
        for (int v = 0; v < 4; ++v) {
            const int col = vh * 64 + v * 16 + q;
            Out[(rowbase + mrow) * 128 + col] = o1[v][r] * il1 - lam * o2[v][r] * il2;
        }
    }
}

// ---------------------------------------------------------------------------
extern "C" void kernel_launch(void* const* d_in, const int* in_sizes, int n_in,
                              void* d_out, int out_size, void* d_ws, size_t ws_size,
                              hipStream_t stream) {
    const float* X   = (const float*)d_in[0];   // [2,4096,1024]
    const float* Wq  = (const float*)d_in[1];   // [1024,128]
    const float* Wk  = (const float*)d_in[2];
    const float* Wv  = (const float*)d_in[3];
    const float* lam = (const float*)d_in[4];   // scalar
    float* Out = (float*)d_out;                 // [2,4096,128]

    // Workspace (bf16): Qb 2MB | Kb 2MB | Vt 2MB | Wt 768KB
    __bf16* Qb = (__bf16*)d_ws;
    __bf16* Kb = Qb + (size_t)8192 * 128;
    __bf16* Vt = Kb + (size_t)8192 * 128;
    __bf16* Wt = Vt + (size_t)2 * 128 * 4096;

    // Kernel 0: pack weights column-major bf16 (384*1024 elems / 256)
    DiffAttn_packW<<<1536, 256, 0, stream>>>(Wq, Wk, Wv, Wt);

    // Kernel 1: 512 row-tiles x 24 col-tiles = 12288 waves; 8 waves/block
    DiffAttn_qkv_proj<<<1536, 256, 0, stream>>>(X, Wt, Qb, Kb, Vt);

    // Kernel 2: 2 batches x 256 qtiles x 2 v-halves = 1024 waves; 8 waves/block
    DiffAttn_flash<<<128, 256, 0, stream>>>(Qb, Kb, Vt, lam, Out);
}
